// GrokOneTransformer_46617575031312
// MI455X (gfx1250) — matrix operand order repaired and checked by reading, not compiled, not run on hardware
//
#include <hip/hip_runtime.h>
#include <hip/hip_bf16.h>
#include <cmath>

#define S_  2048
#define D_  1024
#define H_  16
#define DK_ 64
#define E_  8
#define FF_ 4096

typedef __attribute__((ext_vector_type(16))) _Float16 v16h;
typedef __attribute__((ext_vector_type(8)))  _Float16 v8h;
typedef __attribute__((ext_vector_type(8)))  float    v8f;

static __device__ __forceinline__ v8f zero8() {
    v8f z;
#pragma unroll
    for (int i = 0; i < 8; ++i) z[i] = 0.0f;
    return z;
}

// A fragment (16x32 f16): lane m=lane&15 holds row m.
// elements j -> K = k0 + half*16 + j
static __device__ __forceinline__ v16h afrag(const _Float16* rowk0, int half) {
    return *(const v16h*)(rowk0 + half * 16);
}

// B fragment (32x16 f16): lane n=lane&15 holds col n; elements j -> K = k0 + half*16 + j
static __device__ __forceinline__ v16h bfrag(const _Float16* rowk0, int half) {
    return *(const v16h*)(rowk0 + half * 16);
}

static __device__ __forceinline__ v8f wmma_f16(v16h a, v16h b, v8f c) {
    return __builtin_amdgcn_wmma_f32_16x16x32_f16(false, a, false, b, (short)0, c,
                                                  false, false);
}

static __device__ __forceinline__ float redmax16(float v) {
#pragma unroll
    for (int off = 1; off < 16; off <<= 1) v = fmaxf(v, __shfl_xor(v, off, 32));
    return v;
}
static __device__ __forceinline__ float redsum16(float v) {
#pragma unroll
    for (int off = 1; off < 16; off <<= 1) v += __shfl_xor(v, off, 32);
    return v;
}

// ---------------------------------------------------------------------------
// Generic WMMA GEMM: C[M,N] = A[M,K](f16,row-major) x B[N,K](f16,row-major, i.e. B^T)
// mode 0: Cf[m,n]  = acc
// mode 1: Cf[m,n] += (row_scale ? row_scale[m*sstride] : 1) * acc
// mode 2: Ch[m,n]  = (f16)acc
// Wave computes 32(M) x 64(N) (8 WMMA tiles; B frags reused across 2 A rows).
// Block = 8 waves arranged 4(M) x 2(N) -> 128 x 128 block tile.
// ---------------------------------------------------------------------------
__global__ __launch_bounds__(256) void gemm_wmma_kernel(
    const _Float16* __restrict__ A, int lda,
    const _Float16* __restrict__ B, int ldb,
    float* __restrict__ Cf, _Float16* __restrict__ Ch, int ldc,
    int M, int N, int K,
    const float* __restrict__ row_scale, int sstride, int mode)
{
    const int lane = threadIdx.x & 31;
    const int wave = threadIdx.x >> 5;       // 0..7
    const int nn   = lane & 15;
    const int half = lane >> 4;
    const int mbase = blockIdx.y * 128 + (wave >> 1) * 32;
    const int nbase = blockIdx.x * 128 + (wave & 1) * 64;
    if (mbase >= M || nbase >= N) return;    // wave-uniform: EXEC stays all-ones

    v8f acc[2][4];
#pragma unroll
    for (int t = 0; t < 2; ++t)
#pragma unroll
        for (int j = 0; j < 4; ++j) acc[t][j] = zero8();

    const _Float16* arow0 = A + (size_t)(mbase + nn) * lda;
    const _Float16* arow1 = A + (size_t)(mbase + 16 + nn) * lda;
    const _Float16* brow[4] = {
        B + (size_t)(nbase +  0 + nn) * ldb,
        B + (size_t)(nbase + 16 + nn) * ldb,
        B + (size_t)(nbase + 32 + nn) * ldb,
        B + (size_t)(nbase + 48 + nn) * ldb,
    };

    for (int k0 = 0; k0 < K; k0 += 32) {
        v16h a0 = afrag(arow0 + k0, half);
        v16h a1 = afrag(arow1 + k0, half);
#pragma unroll
        for (int j = 0; j < 4; ++j) {
            v16h b = bfrag(brow[j] + k0, half);
            acc[0][j] = wmma_f16(a0, b, acc[0][j]);
            acc[1][j] = wmma_f16(a1, b, acc[1][j]);
        }
    }

#pragma unroll
    for (int t = 0; t < 2; ++t) {
#pragma unroll
        for (int j = 0; j < 4; ++j) {
            const int n = nbase + j * 16 + nn;
#pragma unroll
            for (int i = 0; i < 8; ++i) {
                const int m = mbase + t * 16 + i + 8 * half;
                const size_t off = (size_t)m * ldc + n;
                if (mode == 0) {
                    Cf[off] = acc[t][j][i];
                } else if (mode == 1) {
                    float sc = row_scale ? row_scale[(size_t)m * sstride] : 1.0f;
                    Cf[off] += sc * acc[t][j][i];
                } else {
                    Ch[off] = (_Float16)acc[t][j][i];
                }
            }
        }
    }
}

// ---------------------------------------------------------------------------
// RMSNorm: out_f16[s,d] = scale[d] * x[s,d] / sqrt(mean(x^2) + eps)
// ---------------------------------------------------------------------------
__global__ __launch_bounds__(256) void rmsnorm_kernel(
    const float* __restrict__ x, const float* __restrict__ scale,
    _Float16* __restrict__ out)
{
    const int s = blockIdx.x;
    const int t = threadIdx.x;
    __shared__ float red[256];
    float acc = 0.f;
    for (int d = t; d < D_; d += 256) {
        float v = x[(size_t)s * D_ + d];
        acc += v * v;
    }
    red[t] = acc;
    __syncthreads();
    for (int off = 128; off > 0; off >>= 1) {
        if (t < off) red[t] += red[t + off];
        __syncthreads();
    }
    const float inv = rsqrtf(red[0] / (float)D_ + 1e-5f);
    for (int d = t; d < D_; d += 256)
        out[(size_t)s * D_ + d] = (_Float16)(scale[d] * x[(size_t)s * D_ + d] * inv);
}

__global__ void f32_to_f16_kernel(const float* __restrict__ in,
                                  _Float16* __restrict__ out, int n)
{
    int i = blockIdx.x * 256 + threadIdx.x;
    if (i < n) out[i] = (_Float16)in[i];
}

// vT[d, t] = vf[t, d]   (d in [0,1024), t in [0,2048))
__global__ void transpose_v_kernel(const _Float16* __restrict__ vf,
                                   _Float16* __restrict__ vT)
{
    int i = blockIdx.x * 256 + threadIdx.x;
    if (i < D_ * S_) {
        int d = i >> 11;       // /2048
        int t = i & (S_ - 1);
        vT[i] = vf[(size_t)t * D_ + d];
    }
}

// ---------------------------------------------------------------------------
// Flash attention, causal.  1 wave per block; grid = (S/16, H).
// q/k f16 [S, H*DK]; vT f16 [H*DK, S]; ctx f16 [S, H*DK].
// ---------------------------------------------------------------------------
__global__ __launch_bounds__(32) void attn_kernel(
    const _Float16* __restrict__ qf, const _Float16* __restrict__ kf,
    const _Float16* __restrict__ vT, _Float16* __restrict__ ctx)
{
    __shared__ _Float16 pls[16 * 32];
    const int qb   = blockIdx.x * 16;
    const int h    = blockIdx.y;
    const int lane = threadIdx.x;
    const int nn   = lane & 15;
    const int half = lane >> 4;

    const _Float16* qrow = qf + (size_t)(qb + nn) * D_ + h * DK_;
    const v16h aq0 = afrag(qrow, half);       // q dims 0..31
    const v16h aq1 = afrag(qrow + 32, half);  // q dims 32..63

    float m[8], l[8], al[8];
#pragma unroll
    for (int i = 0; i < 8; ++i) { m[i] = -INFINITY; l[i] = 0.f; }
    v8f o0 = zero8(), o1 = zero8(), o2 = zero8(), o3 = zero8();

    for (int kb = 0; kb < qb + 16; kb += 32) {
        // --- scores: two 16-col subtiles ---
        v8f s0 = zero8(), s1 = zero8();
        {
            const _Float16* kr0 = kf + (size_t)(kb + nn) * D_ + h * DK_;
            s0 = wmma_f16(aq0, bfrag(kr0, half), s0);
            s0 = wmma_f16(aq1, bfrag(kr0 + 32, half), s0);
            const _Float16* kr1 = kf + (size_t)(kb + 16 + nn) * D_ + h * DK_;
            s1 = wmma_f16(aq0, bfrag(kr1, half), s1);
            s1 = wmma_f16(aq1, bfrag(kr1 + 32, half), s1);
        }
        // --- online softmax update ---
#pragma unroll
        for (int i = 0; i < 8; ++i) {
            const int row = qb + i + 8 * half;
            const int c0 = kb + nn, c1 = kb + 16 + nn;
            float a0 = (c0 <= row) ? s0[i] * 0.125f : -INFINITY;
            float a1 = (c1 <= row) ? s1[i] * 0.125f : -INFINITY;
            float tm = redmax16(fmaxf(a0, a1));
            float mn = fmaxf(m[i], tm);
            float p0 = expf(a0 - mn);
            float p1 = expf(a1 - mn);
            float rs = redsum16(p0 + p1);
            al[i] = expf(m[i] - mn);
            l[i] = l[i] * al[i] + rs;
            m[i] = mn;
            pls[(i + 8 * half) * 32 + nn] = (_Float16)p0;
            pls[(i + 8 * half) * 32 + 16 + nn] = (_Float16)p1;
        }
#pragma unroll
        for (int i = 0; i < 8; ++i) {
            o0[i] *= al[i]; o1[i] *= al[i]; o2[i] *= al[i]; o3[i] *= al[i];
        }
        asm volatile("s_wait_dscnt 0" ::: "memory");
        // P in A layout (contraction K = 32 key positions of this tile)
        const v16h ap = afrag(&pls[nn * 32], half);
        // V^T B fragments: lane nn = dk within each 16-wide tile; K contiguous in t
        o0 = wmma_f16(ap, *(const v16h*)(vT + (size_t)(h * DK_ +  0 + nn) * S_ + kb + half * 16), o0);
        o1 = wmma_f16(ap, *(const v16h*)(vT + (size_t)(h * DK_ + 16 + nn) * S_ + kb + half * 16), o1);
        o2 = wmma_f16(ap, *(const v16h*)(vT + (size_t)(h * DK_ + 32 + nn) * S_ + kb + half * 16), o2);
        o3 = wmma_f16(ap, *(const v16h*)(vT + (size_t)(h * DK_ + 48 + nn) * S_ + kb + half * 16), o3);
    }

#pragma unroll
    for (int i = 0; i < 8; ++i) {
        const int row = qb + i + 8 * half;
        const float inv = 1.0f / l[i];
        _Float16* crow = ctx + (size_t)row * D_ + h * DK_;
        crow[ 0 + nn] = (_Float16)(o0[i] * inv);
        crow[16 + nn] = (_Float16)(o1[i] * inv);
        crow[32 + nn] = (_Float16)(o2[i] * inv);
        crow[48 + nn] = (_Float16)(o3[i] * inv);
    }
}

// ---------------------------------------------------------------------------
// Router: logits = nrm2 @ Wg^T, softmax(8), top-2 renormalized combine weights.
// One block (256 threads) per token.
// ---------------------------------------------------------------------------
__global__ __launch_bounds__(256) void router_kernel(
    const _Float16* __restrict__ nrm2, const float* __restrict__ Wg,
    float* __restrict__ probs, float* __restrict__ combine)
{
    const int s = blockIdx.x;
    const int t = threadIdx.x;
    __shared__ float red[256];
    __shared__ float logits[E_];
    float xl[4];
#pragma unroll
    for (int j = 0; j < 4; ++j) xl[j] = (float)nrm2[(size_t)s * D_ + t * 4 + j];
    for (int e = 0; e < E_; ++e) {
        const float* w = Wg + (size_t)e * D_ + t * 4;
        red[t] = xl[0] * w[0] + xl[1] * w[1] + xl[2] * w[2] + xl[3] * w[3];
        __syncthreads();
        for (int off = 128; off > 0; off >>= 1) {
            if (t < off) red[t] += red[t + off];
            __syncthreads();
        }
        if (t == 0) logits[e] = red[0];
        __syncthreads();
    }
    if (t == 0) {
        float mx = logits[0];
        for (int e = 1; e < E_; ++e) mx = fmaxf(mx, logits[e]);
        float p[E_], sum = 0.f;
        for (int e = 0; e < E_; ++e) { p[e] = expf(logits[e] - mx); sum += p[e]; }
        for (int e = 0; e < E_; ++e) {
            p[e] /= sum;
            probs[(size_t)s * E_ + e] = p[e];
        }
        int i0 = 0;
        for (int e = 1; e < E_; ++e) if (p[e] > p[i0]) i0 = e;
        int i1 = (i0 == 0) ? 1 : 0;
        for (int e = 0; e < E_; ++e) if (e != i0 && p[e] > p[i1]) i1 = e;
        const float denom = p[i0] + p[i1];
        for (int e = 0; e < E_; ++e) combine[(size_t)s * E_ + e] = 0.f;
        combine[(size_t)s * E_ + i0] = p[i0] / denom;
        combine[(size_t)s * E_ + i1] = p[i1] / denom;
    }
}

// h = f16( gelu_exact(g) * v )
__global__ void gelu_mul_kernel(const float* __restrict__ g,
                                const float* __restrict__ v,
                                _Float16* __restrict__ h, int n)
{
    int i = blockIdx.x * 256 + threadIdx.x;
    if (i < n) {
        float x = g[i];
        float ge = 0.5f * x * (1.0f + erff(x * 0.70710678118654752f));
        h[i] = (_Float16)(ge * v[i]);
    }
}

// ---------------------------------------------------------------------------
extern "C" void kernel_launch(void* const* d_in, const int* in_sizes, int n_in,
                              void* d_out, int out_size, void* d_ws, size_t ws_size,
                              hipStream_t stream)
{
    (void)in_sizes; (void)n_in; (void)out_size; (void)ws_size;
    const float* x          = (const float*)d_in[0];
    // d_in[1] = self_attn_mask (causality implemented in-kernel)
    const float* scale_attn = (const float*)d_in[2];
    const float* Wq         = (const float*)d_in[3];
    const float* Wk         = (const float*)d_in[4];
    const float* Wv         = (const float*)d_in[5];
    const float* Wo         = (const float*)d_in[6];
    const float* scale_ffn  = (const float*)d_in[7];
    const float* Wg         = (const float*)d_in[8];
    const float* W_in       = (const float*)d_in[9];
    const float* W_v        = (const float*)d_in[10];
    const float* W_out      = (const float*)d_in[11];

    float* out   = (float*)d_out;                 // [S, D]
    float* probs = out + (size_t)S_ * D_;         // [S, E]

    // ---- workspace carve-up (256B aligned) ----
    char* ws = (char*)d_ws;
    size_t p = 0;
    auto carve = [&](size_t bytes) {
        char* r = ws + p;
        p = (p + bytes + 255) & ~(size_t)255;
        return r;
    };
    const size_t SD = (size_t)S_ * D_;
    const size_t SF = (size_t)S_ * FF_;
    _Float16* nrm1   = (_Float16*)carve(SD * 2);
    _Float16* qf     = (_Float16*)carve(SD * 2);
    _Float16* kf     = (_Float16*)carve(SD * 2);
    _Float16* vf     = (_Float16*)carve(SD * 2);
    _Float16* vT     = (_Float16*)carve(SD * 2);
    _Float16* ctx    = (_Float16*)carve(SD * 2);
    _Float16* wbuf   = (_Float16*)carve((size_t)FF_ * D_ * 2);   // reused weight buffer
    float*    x1     = (float*)   carve(SD * 4);
    _Float16* nrm2   = (_Float16*)carve(SD * 2);
    float*    comb   = (float*)   carve((size_t)S_ * E_ * 4);
    float*    gbuf   = (float*)   carve(SF * 4);
    float*    vbuf   = (float*)   carve(SF * 4);
    _Float16* hbuf   = (_Float16*)carve(SF * 2);

    const dim3 blk256(256);
    const dim3 gemmG_DD(D_ / 128, S_ / 128);    // N=1024
    const dim3 gemmG_FF(FF_ / 128, S_ / 128);   // N=4096
    const int  convDD = (D_ * D_ + 255) / 256;
    const int  convFD = (FF_ * D_ + 255) / 256;

    // 1) attention pre-norm
    rmsnorm_kernel<<<S_, blk256, 0, stream>>>(x, scale_attn, nrm1);

    // 2) Q/K/V projections (Wq/Wk/Wv flat: [H*DK, D] row-major == [N,K])
    f32_to_f16_kernel<<<convDD, blk256, 0, stream>>>(Wq, wbuf, D_ * D_);
    gemm_wmma_kernel<<<gemmG_DD, blk256, 0, stream>>>(nrm1, D_, wbuf, D_,
        nullptr, qf, D_, S_, D_, D_, nullptr, 0, 2);
    f32_to_f16_kernel<<<convDD, blk256, 0, stream>>>(Wk, wbuf, D_ * D_);
    gemm_wmma_kernel<<<gemmG_DD, blk256, 0, stream>>>(nrm1, D_, wbuf, D_,
        nullptr, kf, D_, S_, D_, D_, nullptr, 0, 2);
    f32_to_f16_kernel<<<convDD, blk256, 0, stream>>>(Wv, wbuf, D_ * D_);
    gemm_wmma_kernel<<<gemmG_DD, blk256, 0, stream>>>(nrm1, D_, wbuf, D_,
        nullptr, vf, D_, S_, D_, D_, nullptr, 0, 2);

    // 3) transpose V -> [H*DK, S] for contiguous P·V fragments
    transpose_v_kernel<<<(int)((SD + 255) / 256), blk256, 0, stream>>>(vf, vT);

    // 4) flash attention
    attn_kernel<<<dim3(S_ / 16, H_), dim3(32), 0, stream>>>(qf, kf, vT, ctx);

    // 5) x1 = x + ctx @ Wo^T
    hipMemcpyAsync(x1, x, SD * sizeof(float), hipMemcpyDeviceToDevice, stream);
    f32_to_f16_kernel<<<convDD, blk256, 0, stream>>>(Wo, wbuf, D_ * D_);
    gemm_wmma_kernel<<<gemmG_DD, blk256, 0, stream>>>(ctx, D_, wbuf, D_,
        x1, nullptr, D_, S_, D_, D_, nullptr, 0, 1);

    // 6) FFN pre-norm + router
    rmsnorm_kernel<<<S_, blk256, 0, stream>>>(x1, scale_ffn, nrm2);
    router_kernel<<<S_, blk256, 0, stream>>>(nrm2, Wg, probs, comb);

    // 7) out = x1 + sum_e combine[:,e] * expert_e(nrm2)
    hipMemcpyAsync(out, x1, SD * sizeof(float), hipMemcpyDeviceToDevice, stream);
    for (int e = 0; e < E_; ++e) {
        const size_t woff = (size_t)e * FF_ * D_;
        f32_to_f16_kernel<<<convFD, blk256, 0, stream>>>(W_in + woff, wbuf, FF_ * D_);
        gemm_wmma_kernel<<<gemmG_FF, blk256, 0, stream>>>(nrm2, D_, wbuf, D_,
            gbuf, nullptr, FF_, S_, FF_, D_, nullptr, 0, 0);
        f32_to_f16_kernel<<<convFD, blk256, 0, stream>>>(W_v + woff, wbuf, FF_ * D_);
        gemm_wmma_kernel<<<gemmG_FF, blk256, 0, stream>>>(nrm2, D_, wbuf, D_,
            vbuf, nullptr, FF_, S_, FF_, D_, nullptr, 0, 0);
        gelu_mul_kernel<<<(int)((SF + 255) / 256), blk256, 0, stream>>>(
            gbuf, vbuf, hbuf, (int)SF);
        f32_to_f16_kernel<<<convFD, blk256, 0, stream>>>(W_out + woff, wbuf, D_ * FF_);
        gemm_wmma_kernel<<<gemmG_DD, blk256, 0, stream>>>(hbuf, FF_, wbuf, FF_,
            out, nullptr, D_, S_, D_, FF_, comb + e, E_, 1);
    }
}
